// MultiHeadAttention_84078279787190
// MI455X (gfx1250) — compile-verified, matrix operand order checked by reading, not run on hardware
//
#include <hip/hip_runtime.h>

// ---------------------------------------------------------------------------
// MHA forward for MI455X (gfx1250, wave32, WMMA + TDM + async LDS loads).
// out = (ctx @ Wo + bo), att = softmax(scale * (q@Wq+bq)h @ (k@Wk+bk)h^T)
// bf16 WMMA (v_wmma_f32_16x16x32_bf16) with f32 accumulation.
// ---------------------------------------------------------------------------

typedef __attribute__((ext_vector_type(16))) __bf16 v16bf;
typedef __attribute__((ext_vector_type(8)))  float  v8f;
typedef __attribute__((ext_vector_type(4)))  unsigned int v4u;
typedef __attribute__((ext_vector_type(8)))  unsigned int v8u;

union BfFrag {
  v16bf bf;
  unsigned int u[8];
};

__device__ __forceinline__ unsigned short f2bf(float f) {
  unsigned int u = __float_as_uint(f);
  u += 0x7FFFu + ((u >> 16) & 1u);   // round-to-nearest-even
  return (unsigned short)(u >> 16);
}

// K offset inside a 32-wide step for fragment VGPR v (0..7), lane-group L.
__device__ __forceinline__ int frag_k(int v, int L) {
  return ((v < 4) ? (2 * v) : (16 + 2 * (v - 4))) + 8 * L;
}

// Load a 16(row)x32(k) bf16 fragment from LDS (row-major ushort, even stride).
__device__ __forceinline__ v16bf load_frag(const unsigned short* base, int row0,
                                           int sa, int kk) {
  const int lane = (int)(threadIdx.x & 31u);
  const int L = lane >> 4;
  const int r = lane & 15;
  const unsigned short* rp = base + (row0 + r) * sa + kk;
  BfFrag f;
#pragma unroll
  for (int v = 0; v < 8; ++v) {
    f.u[v] = *(const unsigned int*)(rp + frag_k(v, L));
  }
  return f.bf;
}

__device__ __forceinline__ v8f wmma_bf16(v16bf a, v16bf b, v8f c) {
  return __builtin_amdgcn_wmma_f32_16x16x32_bf16(
      false, a, false, b, (short)0, c, false, false);
}

// --------------------------- CDNA5 data movers -----------------------------

// TDM: DMA a 2D bf16 tile (tile_d0 x tile_d1 elements) from global to LDS,
// with optional per-row LDS padding. Issued once per calling wave (EXEC
// ignored); caller must s_wait_tensorcnt + barrier before consumers read.
// Descriptor layout per CDNA5 ISA.
__device__ __forceinline__ void tdm_load_2d_bf16(
    unsigned lds_addr, const void* gaddr, unsigned tensor_d0,
    unsigned tensor_d1, unsigned tile_d0, unsigned tile_d1,
    unsigned row_stride, unsigned pad_interval_code, unsigned pad_amount_code,
    unsigned pad_en) {
  unsigned long long ga = (unsigned long long)gaddr;
  v4u g0;
  g0[0] = 1u;                                     // count=1 (valid descriptor)
  g0[1] = lds_addr;                               // LDS byte address
  g0[2] = (unsigned)(ga & 0xFFFFFFFFu);           // global_addr[31:0]
  g0[3] = (unsigned)((ga >> 32) & 0x01FFFFFFu)    // global_addr[56:32]
          | (2u << 30);                           // type=2 ("image")
  v8u g1;
  g1[0] = (1u << 16)                              // data_size=1 -> 2 bytes
          | (pad_en << 20)                        // pad_enable
          | (pad_interval_code << 22)             // DWORDs between pads
          | (pad_amount_code << 25);              // pad DWORDs - 1
  g1[1] = (tensor_d0 & 0xFFFFu) << 16;            // tensor_dim0[15:0]
  g1[2] = ((tensor_d0 >> 16) & 0xFFFFu) | ((tensor_d1 & 0xFFFFu) << 16);
  g1[3] = ((tensor_d1 >> 16) & 0xFFFFu) | ((tile_d0 & 0xFFFFu) << 16);
  g1[4] = tile_d1 & 0xFFFFu;                      // tile_dim1, tile_dim2=0
  g1[5] = row_stride;                             // tensor_dim0_stride[31:0]
  g1[6] = 0u;
  g1[7] = 0u;
  asm volatile("tensor_load_to_lds %0, %1" :: "s"(g0), "s"(g1) : "memory");
}

// Per-lane async copy of 16 bytes global -> LDS (raw, no conversion).
__device__ __forceinline__ void async_load_b128(unsigned lds_byte_addr,
                                                const void* gaddr) {
  asm volatile("global_load_async_to_lds_b128 %0, %1, off"
               :: "v"(lds_byte_addr), "v"(gaddr) : "memory");
}
__device__ __forceinline__ void wait_async0() {
  asm volatile("s_wait_asynccnt 0x0" ::: "memory");
}

__device__ __forceinline__ unsigned lds_off(const void* p) {
  // Flat LDS-aperture addresses carry the workgroup-relative LDS byte offset
  // in their low 32 bits (CDNA5 aperture mapping).
  return (unsigned)(unsigned long long)p;
}

// ---------------------------------------------------------------------------
// Projection: Y = X(f32, MxK) @ W(f32, KxN) + bias, stored bf16 head-split:
// Yp[((b*H + h)*S + s)*D + d].  M=4096, K=N=1024.  Block 128x128, 8 waves.
// ---------------------------------------------------------------------------
__global__ __launch_bounds__(256)
void proj_qkv_kernel(const float* __restrict__ X, const float* __restrict__ W,
                     const float* __restrict__ bias,
                     unsigned short* __restrict__ Yp) {
  const int K = 1024, N = 1024;
  __shared__ __align__(16) unsigned short As[128 * 40];
  __shared__ __align__(16) unsigned short Bs[128 * 40];  // transposed: [n][k]
  const int t = (int)threadIdx.x;
  const int wave = t >> 5;
  const int m0 = (int)blockIdx.y * 128;
  const int n0 = (int)blockIdx.x * 128;
  const int wr = wave >> 1, wc = wave & 1;

  const v8f vzero = {0.f, 0.f, 0.f, 0.f, 0.f, 0.f, 0.f, 0.f};
  v8f acc[2][4];
#pragma unroll
  for (int i = 0; i < 2; ++i)
#pragma unroll
    for (int j = 0; j < 4; ++j) acc[i][j] = vzero;

  for (int k0 = 0; k0 < K; k0 += 32) {
#pragma unroll
    for (int i = 0; i < 16; ++i) {  // A tile 128x32, f32 -> bf16
      int idx = t + i * 256;
      int r = idx >> 5, c = idx & 31;
      As[r * 40 + c] = f2bf(X[(m0 + r) * K + k0 + c]);
    }
#pragma unroll
    for (int i = 0; i < 16; ++i) {  // B tile 32x128 -> BsT[n][k]
      int idx = t + i * 256;
      int r = idx >> 7, c = idx & 127;
      Bs[c * 40 + r] = f2bf(W[(k0 + r) * N + n0 + c]);
    }
    __syncthreads();

    v16bf af[2], bf[4];
#pragma unroll
    for (int i = 0; i < 2; ++i) af[i] = load_frag(As, wr * 32 + i * 16, 40, 0);
#pragma unroll
    for (int j = 0; j < 4; ++j) bf[j] = load_frag(Bs, wc * 64 + j * 16, 40, 0);
#pragma unroll
    for (int i = 0; i < 2; ++i)
#pragma unroll
      for (int j = 0; j < 4; ++j) acc[i][j] = wmma_bf16(af[i], bf[j], acc[i][j]);
    __syncthreads();
  }

  const int lane = t & 31;
  const int L = lane >> 4;
  const int nc = lane & 15;
#pragma unroll
  for (int i = 0; i < 2; ++i)
#pragma unroll
    for (int j = 0; j < 4; ++j)
#pragma unroll
      for (int r = 0; r < 8; ++r) {
        int mg = m0 + wr * 32 + i * 16 + L * 8 + r;
        int ng = n0 + wc * 64 + j * 16 + nc;
        float val = acc[i][j][r] + bias[ng];
        int b = mg >> 10, s = mg & 1023;
        int h = ng >> 6, d = ng & 63;
        Yp[(((b * 16 + h) << 10) + s) * 64 + d] = f2bf(val);
      }
}

// ---------------------------------------------------------------------------
// Output projection: Y(f32) = Xb(bf16, 4096x1024) @ W(f32) + bias.
// A tile is a raw bf16 copy -> use global_load_async_to_lds_b128.
// ---------------------------------------------------------------------------
__global__ __launch_bounds__(256)
void proj_out_kernel(const unsigned short* __restrict__ Xb,
                     const float* __restrict__ W,
                     const float* __restrict__ bias, float* __restrict__ Y) {
  const int K = 1024, N = 1024;
  __shared__ __align__(16) unsigned short As[128 * 40];
  __shared__ __align__(16) unsigned short Bs[128 * 40];
  const int t = (int)threadIdx.x;
  const int wave = t >> 5;
  const int m0 = (int)blockIdx.y * 128;
  const int n0 = (int)blockIdx.x * 128;
  const int wr = wave >> 1, wc = wave & 1;
  const unsigned as_base = lds_off(As);

  const v8f vzero = {0.f, 0.f, 0.f, 0.f, 0.f, 0.f, 0.f, 0.f};
  v8f acc[2][4];
#pragma unroll
  for (int i = 0; i < 2; ++i)
#pragma unroll
    for (int j = 0; j < 4; ++j) acc[i][j] = vzero;

  for (int k0 = 0; k0 < K; k0 += 32) {
    // A tile 128x32 bf16: 512 x 16B chunks, 2 per thread, async to LDS.
#pragma unroll
    for (int i = 0; i < 2; ++i) {
      int id = t + i * 256;
      int r = id >> 2, cc = id & 3;  // row, 8-elem chunk
      async_load_b128(as_base + (unsigned)(r * 80 + cc * 16),
                      Xb + (m0 + r) * K + k0 + cc * 8);
    }
#pragma unroll
    for (int i = 0; i < 16; ++i) {  // B tile 32x128 -> BsT[n][k] (f32->bf16)
      int idx = t + i * 256;
      int r = idx >> 7, c = idx & 127;
      Bs[c * 40 + r] = f2bf(W[(k0 + r) * N + n0 + c]);
    }
    wait_async0();
    __syncthreads();

    v16bf af[2], bf[4];
#pragma unroll
    for (int i = 0; i < 2; ++i) af[i] = load_frag(As, wr * 32 + i * 16, 40, 0);
#pragma unroll
    for (int j = 0; j < 4; ++j) bf[j] = load_frag(Bs, wc * 64 + j * 16, 40, 0);
#pragma unroll
    for (int i = 0; i < 2; ++i)
#pragma unroll
      for (int j = 0; j < 4; ++j) acc[i][j] = wmma_bf16(af[i], bf[j], acc[i][j]);
    __syncthreads();
  }

  const int lane = t & 31;
  const int L = lane >> 4;
  const int nc = lane & 15;
#pragma unroll
  for (int i = 0; i < 2; ++i)
#pragma unroll
    for (int j = 0; j < 4; ++j)
#pragma unroll
      for (int r = 0; r < 8; ++r) {
        int mg = m0 + wr * 32 + i * 16 + L * 8 + r;
        int ng = n0 + wc * 64 + j * 16 + nc;
        Y[mg * N + ng] = acc[i][j][r] + bias[ng];
      }
}

// ---------------------------------------------------------------------------
// logits = scale * qh @ kh^T per (b,h).  M=N=1024, K=D=64.
// Q/K tiles (128x64 bf16) are raw copies -> Tensor Data Mover with HW row
// padding (72-ushort LDS stride for bank-conflict-free fragment loads).
// ---------------------------------------------------------------------------
__global__ __launch_bounds__(256)
void attn_logits_kernel(const unsigned short* __restrict__ qh,
                        const unsigned short* __restrict__ kh,
                        float* __restrict__ att) {
  const int S = 1024, D = 64;
  __shared__ __align__(16) unsigned short As[128 * 72];
  __shared__ __align__(16) unsigned short Bs[128 * 72];  // BsT[n][k]=kh[n][k]
  const int t = (int)threadIdx.x;
  const int wave = t >> 5;
  const int bh = (int)blockIdx.z;
  const int m0 = (int)blockIdx.y * 128;
  const int n0 = (int)blockIdx.x * 128;
  const unsigned short* qb = qh + bh * S * D;
  const unsigned short* kb = kh + bh * S * D;

  if (wave == 0) {
    // 128 rows x 64 bf16; pad 4 DWORDs (code 3) after every 32 DWORDs (code 4)
    tdm_load_2d_bf16(lds_off(As), qb + m0 * D, D, S, D, 128, D, 4u, 3u, 1u);
    tdm_load_2d_bf16(lds_off(Bs), kb + n0 * D, D, S, D, 128, D, 4u, 3u, 1u);
    __builtin_amdgcn_s_wait_tensorcnt(0);
  }
  __syncthreads();

  const int wr = wave >> 1, wc = wave & 1;
  const v8f vzero = {0.f, 0.f, 0.f, 0.f, 0.f, 0.f, 0.f, 0.f};
  v8f acc[2][4];
#pragma unroll
  for (int i = 0; i < 2; ++i)
#pragma unroll
    for (int j = 0; j < 4; ++j) acc[i][j] = vzero;

#pragma unroll
  for (int kk = 0; kk < 64; kk += 32) {
    v16bf af[2], bf[4];
#pragma unroll
    for (int i = 0; i < 2; ++i) af[i] = load_frag(As, wr * 32 + i * 16, 72, kk);
#pragma unroll
    for (int j = 0; j < 4; ++j) bf[j] = load_frag(Bs, wc * 64 + j * 16, 72, kk);
#pragma unroll
    for (int i = 0; i < 2; ++i)
#pragma unroll
      for (int j = 0; j < 4; ++j) acc[i][j] = wmma_bf16(af[i], bf[j], acc[i][j]);
  }

  const int lane = t & 31;
  const int L = lane >> 4;
  const int nc = lane & 15;
  const float scale = 0.125f;  // 1/sqrt(64)
#pragma unroll
  for (int i = 0; i < 2; ++i)
#pragma unroll
    for (int j = 0; j < 4; ++j)
#pragma unroll
      for (int r = 0; r < 8; ++r) {
        int mg = m0 + wr * 32 + i * 16 + L * 8 + r;
        int ng = n0 + wc * 64 + j * 16 + nc;
        att[(bh * S + mg) * S + ng] = acc[i][j][r] * scale;
      }
}

// ---------------------------------------------------------------------------
// Row softmax over 1024 columns, in place. One 256-thread block per row.
// ---------------------------------------------------------------------------
__global__ __launch_bounds__(256)
void softmax_kernel(float* __restrict__ att) {
  __shared__ float red[256];
  const int t = (int)threadIdx.x;
  float* p = att + (int)blockIdx.x * 1024;

  float v[4];
  float mx = -3.402823466e38f;
#pragma unroll
  for (int i = 0; i < 4; ++i) {
    v[i] = p[t + i * 256];
    mx = fmaxf(mx, v[i]);
  }
  red[t] = mx;
  __syncthreads();
  for (int s = 128; s > 0; s >>= 1) {
    if (t < s) red[t] = fmaxf(red[t], red[t + s]);
    __syncthreads();
  }
  mx = red[0];
  __syncthreads();

  float sum = 0.f;
#pragma unroll
  for (int i = 0; i < 4; ++i) {
    v[i] = __expf(v[i] - mx);
    sum += v[i];
  }
  red[t] = sum;
  __syncthreads();
  for (int s = 128; s > 0; s >>= 1) {
    if (t < s) red[t] += red[t + s];
    __syncthreads();
  }
  const float inv = 1.0f / red[0];
#pragma unroll
  for (int i = 0; i < 4; ++i) p[t + i * 256] = v[i] * inv;
}

// ---------------------------------------------------------------------------
// ctx = att(f32) @ vh(bf16) per (b,h); M=1024, N=64, K=1024.
// ---------------------------------------------------------------------------
__global__ __launch_bounds__(256)
void attn_ctx_kernel(const float* __restrict__ att,
                     const unsigned short* __restrict__ vh,
                     unsigned short* __restrict__ ctx) {
  const int S = 1024, D = 64;
  __shared__ __align__(16) unsigned short As[128 * 40];
  __shared__ __align__(16) unsigned short Bs[64 * 40];  // BsT[d][k]
  const int t = (int)threadIdx.x;
  const int wave = t >> 5;
  const int bh = (int)blockIdx.y;
  const int m0 = (int)blockIdx.x * 128;
  const float* ab = att + bh * S * S;
  const unsigned short* vb = vh + bh * S * D;

  const v8f vzero = {0.f, 0.f, 0.f, 0.f, 0.f, 0.f, 0.f, 0.f};
  v8f acc[4];
#pragma unroll
  for (int j = 0; j < 4; ++j) acc[j] = vzero;

  for (int k0 = 0; k0 < S; k0 += 32) {
#pragma unroll
    for (int i = 0; i < 16; ++i) {  // att tile 128x32, f32 -> bf16
      int idx = t + i * 256;
      int r = idx >> 5, c = idx & 31;
      As[r * 40 + c] = f2bf(ab[(m0 + r) * S + k0 + c]);
    }
#pragma unroll
    for (int i = 0; i < 8; ++i) {  // vh tile 32x64 -> BsT[d][k]
      int idx = t + i * 256;
      int r = idx >> 6, c = idx & 63;
      Bs[c * 40 + r] = vb[(k0 + r) * D + c];
    }
    __syncthreads();
    v16bf a = load_frag(As, wave * 16, 40, 0);
#pragma unroll
    for (int j = 0; j < 4; ++j) {
      v16bf b = load_frag(Bs, j * 16, 40, 0);
      acc[j] = wmma_bf16(a, b, acc[j]);
    }
    __syncthreads();
  }

  const int lane = t & 31;
  const int L = lane >> 4;
  const int nc = lane & 15;
  const int b = bh >> 4, h = bh & 15;
#pragma unroll
  for (int j = 0; j < 4; ++j)
#pragma unroll
    for (int r = 0; r < 8; ++r) {
      int s = m0 + wave * 16 + L * 8 + r;
      int d = j * 16 + nc;
      ctx[((b * 1024 + s) << 10) + h * 64 + d] = f2bf(acc[j][r]);
    }
}

// ---------------------------------------------------------------------------
extern "C" void kernel_launch(void* const* d_in, const int* in_sizes, int n_in,
                              void* d_out, int out_size, void* d_ws,
                              size_t ws_size, hipStream_t stream) {
  (void)in_sizes; (void)n_in; (void)out_size; (void)ws_size;
  const float* q  = (const float*)d_in[0];
  const float* k  = (const float*)d_in[1];
  const float* v  = (const float*)d_in[2];
  const float* Wq = (const float*)d_in[3];
  const float* bq = (const float*)d_in[4];
  const float* Wk = (const float*)d_in[5];
  const float* bk = (const float*)d_in[6];
  const float* Wv = (const float*)d_in[7];
  const float* bv = (const float*)d_in[8];
  const float* Wo = (const float*)d_in[9];
  const float* bo = (const float*)d_in[10];

  float* out = (float*)d_out;                       // (B,S,E)   4M floats
  float* att = out + (size_t)4 * 1024 * 1024;       // (B,H,S,S) 64M floats

  unsigned short* ws  = (unsigned short*)d_ws;      // bf16 scratch (32 MB)
  unsigned short* qh  = ws;
  unsigned short* kh  = qh + (size_t)4 * 1024 * 1024;
  unsigned short* vh  = kh + (size_t)4 * 1024 * 1024;
  unsigned short* ctx = vh + (size_t)4 * 1024 * 1024;

  const dim3 blk(256);
  const dim3 gproj(8, 32);  // N/128, M/128

  proj_qkv_kernel<<<gproj, blk, 0, stream>>>(q, Wq, bq, qh);
  proj_qkv_kernel<<<gproj, blk, 0, stream>>>(k, Wk, bk, kh);
  proj_qkv_kernel<<<gproj, blk, 0, stream>>>(v, Wv, bv, vh);

  attn_logits_kernel<<<dim3(8, 8, 64), blk, 0, stream>>>(qh, kh, att);
  softmax_kernel<<<dim3(65536), blk, 0, stream>>>(att);
  attn_ctx_kernel<<<dim3(8, 64), blk, 0, stream>>>(att, vh, ctx);

  proj_out_kernel<<<gproj, blk, 0, stream>>>(ctx, Wo, bo, out);
}
